// NeuralHMM_23957327577292
// MI455X (gfx1250) — matrix operand order repaired and checked by reading, not compiled, not run on hardware
//
#include <hip/hip_runtime.h>

// ---------------------------------------------------------------------------
// NeuralHMM forward for MI455X (gfx1250, wave32, WMMA).
// Shapes: B=8 N=64 T=256 D=80 E=512 P=256 H=512 O=512; gates 4H=2048.
// Strategy:
//   1. Pack weights/activations to f16 (K dims padded to mult. of 32).
//   2. Parallel WMMA GEMMs: prenet L1/L2 for all t, LSTM input gates G for
//      all t, U = inputs @ W1e^T (time-invariant half of out_w1).
//   3. Serial LSTM h-recurrence in ONE workgroup (32 waves) using
//      v_wmma_f32_16x16x32_f16 on the h @ w_hh^T GEMM (M=16 pad of B=8),
//      with global_prefetch of G[t] overlapping the WMMA phase.
//   4. Parallel HV = H @ W1h^T, then fused per-(t,b,n-chunk) kernel:
//      hid = relu(HV + U) -> LDS f16 -> WMMA @ out_w2^T -> Gaussian emission.
//   5. Serial HMM alpha recursion: one wave per batch, 2 states/lane,
//      shuffle-based roll + logsumexp (no LDS, no barriers).
// Workspace use ~32.7 MB.
// ---------------------------------------------------------------------------

typedef __attribute__((ext_vector_type(16))) _Float16 v16h;
typedef __attribute__((ext_vector_type(8)))  _Float16 v8h;
typedef __attribute__((ext_vector_type(8)))  float    v8f;

#define Bz 8
#define Nz 64
#define Tz 256
#define Dz 80
#define Ez 512
#define Pz 256
#define Hz 512
#define G4 2048      // 4*H
#define DP 96        // D padded to mult of 32
#define PO 176       // (2D+1)=161 padded to mult of 16
#define NEG_INF (-1000000.0f)
#define LOG2PI 1.8378770664093453f

// ---------------- device math helpers ----------------
__device__ __forceinline__ float sigmf(float x)      { return 1.0f / (1.0f + __expf(-x)); }
__device__ __forceinline__ float logclampf(float x)  { return __logf(fmaxf(x, 1e-4f)); }
__device__ __forceinline__ float softplusf(float x)  { return (x > 20.f) ? x : log1pf(__expf(x)); }
__device__ __forceinline__ float logaddexpf_(float a, float b) {
  float m = fmaxf(a, b);
  return m + log1pf(__expf(-fabsf(a - b)));
}

// ---------------- WMMA fragment loaders (16x16x32 f16) ----------------
// A (16x32 tile of row-major X[ld]): lane L<16 holds row m0+L, K {k0..k0+7, k0+16..k0+23};
// lane L>=16 holds row m0+L-16, K {k0+8..k0+15, k0+24..k0+31}.
__device__ __forceinline__ v16h frag_a(const _Float16* X, int ld, int m0, int k0, int lane) {
  const _Float16* p = X + (size_t)(m0 + (lane & 15)) * ld + k0 + ((lane >> 4) << 3);
  v8h lo = *(const v8h*)p;
  v8h hi = *(const v8h*)(p + 16);
  v16h r;
#pragma unroll
  for (int i = 0; i < 8; ++i) { r[i] = lo[i]; r[i + 8] = hi[i]; }
  return r;
}
// B = W^T (32x16 tile), W row-major [N x K]: lane L holds column n0+(L&15),
// K {k0+8h..k0+8h+7, k0+16+8h..k0+16+8h+7}, h = (L>=16) (same as frag_a).
__device__ __forceinline__ v16h frag_b(const _Float16* W, int ld, int n0, int k0, int lane) {
  const _Float16* p = W + (size_t)(n0 + (lane & 15)) * ld + k0 + ((lane >> 4) << 3);
  v8h lo = *(const v8h*)p;
  v8h hi = *(const v8h*)(p + 16);
  v16h r;
#pragma unroll
  for (int i = 0; i < 8; ++i) { r[i] = lo[i]; r[i + 8] = hi[i]; }
  return r;
}
__device__ __forceinline__ v8f wmma_f16(v16h a, v16h b, v8f c) {
  return __builtin_amdgcn_wmma_f32_16x16x32_f16(false, a, false, b, (short)0, c, false, false);
}

// ---------------- pack: f32 -> f16 with pad / sub-view ----------------
__global__ __launch_bounds__(256) void pack_half_kernel(
    const float* __restrict__ src, int sld, int scol0, int srows, int scols,
    _Float16* __restrict__ dst, int dld, int drows) {
  int i = blockIdx.x * 256 + threadIdx.x;
  int total = drows * dld;
  if (i >= total) return;
  int r = i / dld, c = i % dld;
  float v = (r < srows && c < scols) ? src[(size_t)r * sld + scol0 + c] : 0.f;
  dst[i] = (_Float16)v;
}

// AR inputs: dst[(t*B+b)][DP], d<D: (t==0 ? 0 : mels[b][d][t-1]); mels (B,D,T)
__global__ __launch_bounds__(256) void pack_ar_kernel(const float* __restrict__ mels,
                                                      _Float16* __restrict__ dst) {
  int i = blockIdx.x * 256 + threadIdx.x;
  if (i >= Tz * Bz * DP) return;
  int d = i % DP;
  int tb = i / DP;
  int b = tb % Bz, t = tb / Bz;
  float v = 0.f;
  if (d < Dz && t > 0) v = mels[((size_t)b * Dz + d) * Tz + (t - 1)];
  dst[i] = (_Float16)v;
}

__global__ __launch_bounds__(256) void bias_sum_kernel(const float* __restrict__ a,
                                                       const float* __restrict__ b,
                                                       float* __restrict__ o, int n) {
  int i = blockIdx.x * 256 + threadIdx.x;
  if (i < n) o[i] = a[i] + b[i];
}

// ---------------- generic C = act(X @ W^T + bias) WMMA GEMM ----------------
// X: [M x K] f16 (ldx), W: [N x K] f16 (ldw). M%16==0, N%16==0, K%32==0.
// Cf (f32) and/or Ch (f16) outputs with ld = ldc. One wave per 16x16 tile.
__global__ __launch_bounds__(256) void gemm_xwt_kernel(
    const _Float16* __restrict__ X, int ldx,
    const _Float16* __restrict__ W, int ldw,
    const float* __restrict__ bias,
    float* __restrict__ Cf, _Float16* __restrict__ Ch, int ldc,
    int M, int N, int K, int do_relu) {
  int lane = threadIdx.x & 31, wave = threadIdx.x >> 5;
  int ntilesN = N >> 4;
  int ntiles = (M >> 4) * ntilesN;
  for (int tile = blockIdx.x * 8 + wave; tile < ntiles; tile += gridDim.x * 8) {
    int mt = tile / ntilesN, nt = tile % ntilesN;
    v8f acc = {};
    for (int k0 = 0; k0 < K; k0 += 32) {
      v16h a = frag_a(X, ldx, mt << 4, k0, lane);
      v16h b = frag_b(W, ldw, nt << 4, k0, lane);
      acc = wmma_f16(a, b, acc);
    }
    int n = (nt << 4) + (lane & 15);
    float bv = bias ? bias[n] : 0.f;
#pragma unroll
    for (int v = 0; v < 8; ++v) {
      int m = (mt << 4) + v + ((lane >> 4) << 3);
      float val = acc[v] + bv;
      if (do_relu) val = fmaxf(val, 0.f);
      if (Cf) Cf[(size_t)m * ldc + n] = val;
      if (Ch) Ch[(size_t)m * ldc + n] = (_Float16)val;
    }
  }
}

// ---------------- serial LSTM recurrence (single workgroup, WMMA) ----------
// gates_t = G[t] + h_{t-1} @ Whh^T ; standard LSTM cell; writes h_t (f16).
__global__ __launch_bounds__(1024) void lstm_kernel(const float* __restrict__ G,
                                                    const _Float16* __restrict__ Whh,
                                                    _Float16* __restrict__ Hout) {
  __shared__ _Float16 sh[16 * Hz];   // h_prev, rows 8..15 = zero pad (16 KB)
  __shared__ _Float16 sg[Bz * G4];   // wmma gate partials as f16 (32 KB)
  int tid = threadIdx.x, lane = tid & 31, wave = tid >> 5;
  for (int i = tid; i < 16 * Hz; i += 1024) sh[i] = (_Float16)0.f;
  float c[4] = {0.f, 0.f, 0.f, 0.f};
  int cb = tid >> 7;          // batch 0..7
  int ck = (tid & 127) << 2;  // 4 consecutive hidden units
  __syncthreads();
  for (int t = 0; t < Tz; ++t) {
    const float* Gt = G + (size_t)t * Bz * G4;
    int base = cb * G4 + ck;
    // warm L2->L0 for the update-phase gate reads (global_prefetch_b8,
    // counter-free) so the WMMA phase hides their latency.
    __builtin_prefetch(Gt + base, 0, 3);
    __builtin_prefetch(Gt + base + Hz, 0, 3);
    __builtin_prefetch(Gt + base + 2 * Hz, 0, 3);
    __builtin_prefetch(Gt + base + 3 * Hz, 0, 3);
    // h_prev(16x512) @ Whh^T(2048x512): 128 N-tiles over 32 waves
#pragma unroll
    for (int j = 0; j < 4; ++j) {
      int nt = (wave << 2) + j;
      v8f acc = {};
      for (int k0 = 0; k0 < Hz; k0 += 32) {
        v16h a = frag_a(sh, Hz, 0, k0, lane);
        v16h b = frag_b(Whh, Hz, nt << 4, k0, lane);
        acc = wmma_f16(a, b, acc);
      }
      if (lane < 16) {  // only batch rows 0..7 (VGPR v -> m = v) are real
#pragma unroll
        for (int v = 0; v < 8; ++v)
          sg[v * G4 + (nt << 4) + lane] = (_Float16)acc[v];
      }
    }
    __syncthreads();
#pragma unroll
    for (int q = 0; q < 4; ++q) {
      int k = ck + q;
      int idx = cb * G4 + k;
      float gi = (float)sg[idx]            + Gt[idx];
      float gf = (float)sg[idx + Hz]       + Gt[idx + Hz];
      float gg = (float)sg[idx + 2 * Hz]   + Gt[idx + 2 * Hz];
      float go = (float)sg[idx + 3 * Hz]   + Gt[idx + 3 * Hz];
      float cn = sigmf(gf) * c[q] + sigmf(gi) * tanhf(gg);
      float hn = sigmf(go) * tanhf(cn);
      c[q] = cn;
      _Float16 hh = (_Float16)hn;
      sh[cb * Hz + k] = hh;
      Hout[((size_t)t * Bz + cb) * Hz + k] = hh;
    }
    __syncthreads();
  }
}

// ---------------- fused hid -> out_w2 -> Gaussian emission -----------------
// one block per (t, b, n-chunk of 32). hid=relu(HV+U) in LDS f16, WMMA GEMM
// against out_w2^T (padded 176x512), then per-state log-likelihood.
__global__ __launch_bounds__(256) void phase2_kernel(
    const float* __restrict__ HV, const float* __restrict__ U,
    const _Float16* __restrict__ W2o, const float* __restrict__ ob2,
    const float* __restrict__ mels, const int* __restrict__ ilen,
    float* __restrict__ EMIT, float* __restrict__ TRANS) {
  __shared__ _Float16 sHid[32 * Hz];   // 32 KB
  __shared__ float    sPar[32 * PO];   // 22 KB
  int blk = blockIdx.x;
  int chunk = blk & 1, b = (blk >> 1) & 7, t = blk >> 4;
  int tid = threadIdx.x, lane = tid & 31, wave = tid >> 5;
  const float* hv = HV + ((size_t)t * Bz + b) * Hz;
  const float* u  = U + ((size_t)b * Nz + chunk * 32) * Hz;
  for (int i = tid; i < 32 * Hz; i += 256) {
    int n = i >> 9, o = i & (Hz - 1);
    sHid[i] = (_Float16)fmaxf(hv[o] + u[(size_t)n * Hz + o], 0.f);
  }
  __syncthreads();
  for (int tile = wave; tile < 2 * (PO >> 4); tile += 8) {   // 22 tiles
    int mt = tile / (PO >> 4), nt = tile % (PO >> 4);
    v8f acc = {};
    for (int k0 = 0; k0 < Hz; k0 += 32) {
      v16h a = frag_a(sHid, Hz, mt << 4, k0, lane);
      v16h bb = frag_b(W2o, Hz, nt << 4, k0, lane);
      acc = wmma_f16(a, bb, acc);
    }
    int p = (nt << 4) + (lane & 15);
    float bv = (p < 2 * Dz + 1) ? ob2[p] : 0.f;
#pragma unroll
    for (int v = 0; v < 8; ++v) {
      int m = (mt << 4) + v + ((lane >> 4) << 3);
      sPar[m * PO + p] = acc[v] + bv;
    }
  }
  __syncthreads();
  if (tid < 32) {
    int n = chunk * 32 + tid;
    const float* pr = &sPar[tid * PO];
    float tv = pr[2 * Dz];
    float lp = 0.f;
    for (int d = 0; d < Dz; ++d) {
      float mean = pr[d];
      float sd = fmaxf(softplusf(pr[Dz + d]), 0.001f);
      float x = mels[((size_t)b * Dz + d) * Tz + t];
      float z = (x - mean) / sd;
      lp += -0.5f * z * z - __logf(sd) - 0.5f * LOG2PI;
    }
    float mask = (n < ilen[b]) ? 1.f : 0.f;
    size_t o = ((size_t)t * Bz + b) * Nz + n;
    EMIT[o] = lp * mask;
    TRANS[o] = tv;
  }
}

// ---------------- serial HMM alpha recursion: 1 wave per batch -------------
// Lane L holds states n0=L and n1=L+32. roll-by-1 via shfl_up; per-batch
// logsumexp via shfl_xor tree. No LDS, no barriers: 8 independent waves.
__global__ __launch_bounds__(256) void hmm_kernel(const float* __restrict__ EMIT,
                                                  const float* __restrict__ TRANS,
                                                  const int* __restrict__ ilen,
                                                  const int* __restrict__ mlen,
                                                  float* __restrict__ out) {
  int tid = threadIdx.x, b = tid >> 5, lane = tid & 31;
  int n0 = lane, n1 = lane + 32;
  int il = ilen[b], ml = mlen[b];
  bool m0 = n0 < il, m1 = n1 < il;
  float a0 = 0.f, a1 = 0.f;                 // prev alpha (carry)
  float sum_log_c = 0.f;
  float la0 = NEG_INF, la1 = NEG_INF, ltv0 = 0.f, ltv1 = 0.f;
  for (int t = 0; t < Tz; ++t) {
    const float* Et = EMIT + ((size_t)t * Bz + b) * Nz;
    const float* Tt = TRANS + ((size_t)t * Bz + b) * Nz;
    float e0 = Et[n0], e1 = Et[n1];
    float tv0 = Tt[n0], tv1 = Tt[n1];
    // staying = prev + log(sig(-tv))
    float st0 = a0 + logclampf(sigmf(-tv0));
    float st1 = a1 + logclampf(sigmf(-tv1));
    // leaving[n] = prev[n-1] + log(sig(tv[n-1]))
    float am1_0 = __shfl_up(a0, 1);
    float tm1_0 = __shfl_up(tv0, 1);
    float a0_31 = __shfl(a0, 31);           // state 31 -> feeds state 32
    float tv0_31 = __shfl(tv0, 31);
    float am1_1u = __shfl_up(a1, 1);
    float tm1_1u = __shfl_up(tv1, 1);
    float am1_1 = (lane == 0) ? a0_31 : am1_1u;
    float tm1_1 = (lane == 0) ? tv0_31 : tm1_1u;
    float lv0 = (n0 == 0) ? NEG_INF : am1_0 + logclampf(sigmf(tm1_0));
    float lv1 = am1_1 + logclampf(sigmf(tm1_1));
    float tt0 = m0 ? logaddexpf_(st0, lv0) : NEG_INF;
    float tt1 = m1 ? logaddexpf_(st1, lv1) : NEG_INF;
    float at0 = (t == 0) ? (((n0 == 0) ? 0.f : NEG_INF) + e0) : (e0 + tt0);
    float at1 = (t == 0) ? (NEG_INF + e1) : (e1 + tt1);
    // logsumexp over the 64 states of this batch (intra-wave)
    float mx = fmaxf(at0, at1);
#pragma unroll
    for (int s = 16; s >= 1; s >>= 1) mx = fmaxf(mx, __shfl_xor(mx, s));
    float sm = __expf(at0 - mx) + __expf(at1 - mx);
#pragma unroll
    for (int s = 16; s >= 1; s >>= 1) sm += __shfl_xor(sm, s);
    float log_c = mx + __logf(sm);
    a0 = at0 - log_c;
    a1 = at1 - log_c;
    if (t < ml) sum_log_c += log_c;
    if (t == ml - 1) { la0 = a0; la1 = a1; ltv0 = tv0; ltv1 = tv1; }
  }
  // final: logsumexp(last_alpha + log_trans_p)
  float v0 = (m0 ? la0 : NEG_INF) + ((n0 == il - 1) ? logclampf(sigmf(ltv0)) : NEG_INF);
  float v1 = (m1 ? la1 : NEG_INF) + ((n1 == il - 1) ? logclampf(sigmf(ltv1)) : NEG_INF);
  float mx = fmaxf(v0, v1);
#pragma unroll
  for (int s = 16; s >= 1; s >>= 1) mx = fmaxf(mx, __shfl_xor(mx, s));
  float sm = __expf(v0 - mx) + __expf(v1 - mx);
#pragma unroll
  for (int s = 16; s >= 1; s >>= 1) sm += __shfl_xor(sm, s);
  if (lane == 0) out[b] = sum_log_c + mx + __logf(sm);
}

// ---------------------------------------------------------------------------
extern "C" void kernel_launch(void* const* d_in, const int* in_sizes, int n_in,
                              void* d_out, int out_size, void* d_ws, size_t ws_size,
                              hipStream_t stream) {
  const float* in_inputs = (const float*)d_in[0];
  const float* in_mels   = (const float*)d_in[1];
  const float* w1   = (const float*)d_in[2];
  const float* w2   = (const float*)d_in[3];
  const float* wih  = (const float*)d_in[4];
  const float* whh  = (const float*)d_in[5];
  const float* bih  = (const float*)d_in[6];
  const float* bhh  = (const float*)d_in[7];
  const float* ow1  = (const float*)d_in[8];
  const float* ob1  = (const float*)d_in[9];
  const float* ow2  = (const float*)d_in[10];
  const float* ob2  = (const float*)d_in[11];
  const int*   ilen = (const int*)d_in[12];
  const int*   mlen = (const int*)d_in[13];
  float* out = (float*)d_out;

  // workspace layout (256B aligned), ~32.7 MB total
  char* base = (char*)d_ws;
  size_t off = 0;
  auto alloc = [&](size_t bytes) { size_t o = off; off += (bytes + 255) & ~(size_t)255; return o; };
  _Float16* hW1p = (_Float16*)(base + alloc((size_t)Pz * DP * 2));
  _Float16* hW2p = (_Float16*)(base + alloc((size_t)Pz * Pz * 2));
  _Float16* hWih = (_Float16*)(base + alloc((size_t)G4 * Pz * 2));
  _Float16* hWhh = (_Float16*)(base + alloc((size_t)G4 * Hz * 2));
  _Float16* hW1h = (_Float16*)(base + alloc((size_t)512 * Hz * 2));
  _Float16* hW1e = (_Float16*)(base + alloc((size_t)512 * Ez * 2));
  _Float16* hW2o = (_Float16*)(base + alloc((size_t)PO * 512 * 2));
  _Float16* hAR  = (_Float16*)(base + alloc((size_t)Tz * Bz * DP * 2));
  _Float16* hP1  = (_Float16*)(base + alloc((size_t)Tz * Bz * Pz * 2));
  _Float16* hP2  = (_Float16*)(base + alloc((size_t)Tz * Bz * Pz * 2));
  _Float16* hIn  = (_Float16*)(base + alloc((size_t)Bz * Nz * Ez * 2));
  _Float16* hH   = (_Float16*)(base + alloc((size_t)Tz * Bz * Hz * 2));
  float* fG    = (float*)(base + alloc((size_t)Tz * Bz * G4 * 4));
  float* fU    = (float*)(base + alloc((size_t)Bz * Nz * 512 * 4));
  float* fHV   = (float*)(base + alloc((size_t)Tz * Bz * 512 * 4));
  float* fBsum = (float*)(base + alloc((size_t)G4 * 4));
  float* fEmit = (float*)(base + alloc((size_t)Tz * Bz * Nz * 4));
  float* fTran = (float*)(base + alloc((size_t)Tz * Bz * Nz * 4));
  (void)ws_size; (void)in_sizes; (void)n_in; (void)out_size;

  auto g256 = [](long n) { return (unsigned)((n + 255) / 256); };

  // 1) pack everything to f16 (zero-padded)
  pack_half_kernel<<<g256((long)Pz * DP), 256, 0, stream>>>(w1, Dz, 0, Pz, Dz, hW1p, DP, Pz);
  pack_half_kernel<<<g256((long)Pz * Pz), 256, 0, stream>>>(w2, Pz, 0, Pz, Pz, hW2p, Pz, Pz);
  pack_half_kernel<<<g256((long)G4 * Pz), 256, 0, stream>>>(wih, Pz, 0, G4, Pz, hWih, Pz, G4);
  pack_half_kernel<<<g256((long)G4 * Hz), 256, 0, stream>>>(whh, Hz, 0, G4, Hz, hWhh, Hz, G4);
  pack_half_kernel<<<g256((long)512 * Hz), 256, 0, stream>>>(ow1, Hz + Ez, 0, 512, Hz, hW1h, Hz, 512);
  pack_half_kernel<<<g256((long)512 * Ez), 256, 0, stream>>>(ow1, Hz + Ez, Hz, 512, Ez, hW1e, Ez, 512);
  pack_half_kernel<<<g256((long)PO * 512), 256, 0, stream>>>(ow2, 512, 0, 2 * Dz + 1, 512, hW2o, 512, PO);
  pack_half_kernel<<<g256((long)Bz * Nz * Ez), 256, 0, stream>>>(in_inputs, Ez, 0, Bz * Nz, Ez, hIn, Ez, Bz * Nz);
  pack_ar_kernel<<<g256((long)Tz * Bz * DP), 256, 0, stream>>>(in_mels, hAR);
  bias_sum_kernel<<<g256(G4), 256, 0, stream>>>(bih, bhh, fBsum, G4);

  // 2) parallel WMMA GEMMs
  // P1 = relu(AR @ W1p^T)           [2048 x 256], K=96
  {
    int M = Tz * Bz, N = Pz, K = DP;
    int nt = (M / 16) * (N / 16);
    gemm_xwt_kernel<<<(nt + 7) / 8, 256, 0, stream>>>(hAR, DP, hW1p, DP, nullptr,
                                                      nullptr, hP1, N, M, N, K, 1);
  }
  // P2 = relu(P1 @ W2p^T)           [2048 x 256], K=256
  {
    int M = Tz * Bz, N = Pz, K = Pz;
    int nt = (M / 16) * (N / 16);
    gemm_xwt_kernel<<<(nt + 7) / 8, 256, 0, stream>>>(hP1, Pz, hW2p, Pz, nullptr,
                                                      nullptr, hP2, N, M, N, K, 1);
  }
  // G = P2 @ Wih^T + (b_ih+b_hh)    [2048 x 2048], K=256
  {
    int M = Tz * Bz, N = G4, K = Pz;
    int nt = (M / 16) * (N / 16);
    gemm_xwt_kernel<<<(nt + 7) / 8, 256, 0, stream>>>(hP2, Pz, hWih, Pz, fBsum,
                                                      fG, nullptr, N, M, N, K, 0);
  }
  // U = inputs @ W1e^T + out_b1     [512 x 512], K=512 (time-invariant hoist)
  {
    int M = Bz * Nz, N = 512, K = Ez;
    int nt = (M / 16) * (N / 16);
    gemm_xwt_kernel<<<(nt + 7) / 8, 256, 0, stream>>>(hIn, Ez, hW1e, Ez, ob1,
                                                      fU, nullptr, N, M, N, K, 0);
  }

  // 3) serial LSTM recurrence (WMMA, one workgroup of 32 waves)
  lstm_kernel<<<1, 1024, 0, stream>>>(fG, hWhh, hH);

  // 4) HV = H @ W1h^T               [2048 x 512], K=512
  {
    int M = Tz * Bz, N = 512, K = Hz;
    int nt = (M / 16) * (N / 16);
    gemm_xwt_kernel<<<(nt + 7) / 8, 256, 0, stream>>>(hH, Hz, hW1h, Hz, nullptr,
                                                      fHV, nullptr, N, M, N, K, 0);
  }
  // fused hid -> params -> emission/trans
  phase2_kernel<<<Tz * Bz * 2, 256, 0, stream>>>(fHV, fU, hW2o, ob2, in_mels, ilen,
                                                 fEmit, fTran);

  // 5) serial HMM alpha recursion + final logsumexp (barrier-free)
  hmm_kernel<<<1, 256, 0, stream>>>(fEmit, fTran, ilen, mlen, out);
}
